// NodeAttention_26293789786366
// MI455X (gfx1250) — compile-verified, matrix operand order checked by reading, not run on hardware
//
#include <hip/hip_runtime.h>
#include <hip/hip_bf16.h>

// ---------------------------------------------------------------------------
// NodeAttention for MI455X (gfx1250, wave32, WMMA + TDM).
//   DIM=128, DH=64, H=8, INNER=512, EDIM=32, n=2048, b=1
// k0 weight prep -> k1 LN+QKV (WMMA bf16) -> k2 fused edgeLN+bias+flash
// attention (TDM double-buffered edge stream, bias as WMMA C operand) ->
// k3 output projection (WMMA bf16).
// ---------------------------------------------------------------------------

#define N_TOK   2048
#define DIMC    128
#define DHC     64
#define NH      8
#define INNERC  512
#define EDIMC   32
#define QKV3    1536
#define SCALE_F 0.125f     // DH^-0.5
#define EPS_F   1e-5f

typedef __attribute__((ext_vector_type(16))) __bf16 v16bf;
typedef __attribute__((ext_vector_type(8)))  __bf16 v8bf;
typedef __attribute__((ext_vector_type(8)))  float  v8f;

#if __has_builtin(__builtin_amdgcn_tensor_load_to_lds) && \
    __has_builtin(__builtin_amdgcn_s_wait_tensorcnt)
#define NA_USE_TDM 1
#else
#define NA_USE_TDM 0
#endif

static __device__ __forceinline__ v16bf mk16(v8bf lo, v8bf hi) {
  v16bf r;
#pragma unroll
  for (int i = 0; i < 8; ++i) { r[i] = lo[i]; r[i + 8] = hi[i]; }
  return r;
}
static __device__ __forceinline__ v8bf ld8(const __bf16* p) {
  return *(const v8bf*)p;
}
static __device__ __forceinline__ v8f wmma_bf16(v16bf a, v16bf b, v8f c) {
  // D = A(16x32 bf16) * B(32x16 bf16) + C(16x16 f32)
  return __builtin_amdgcn_wmma_f32_16x16x32_bf16(
      /*neg_a=*/false, a, /*neg_b=*/false, b,
      /*c_mod=*/(short)0, c, /*reuse_a=*/false, /*reuse_b=*/false);
}

#if NA_USE_TDM
typedef __attribute__((ext_vector_type(4))) unsigned int v4u_;
typedef __attribute__((ext_vector_type(8))) int          v8i_;
typedef __attribute__((ext_vector_type(4))) int          v4i_;

// TDM: DMA a 2-D tile (16 rows x 1024 f32, row stride 65536 f32) from global
// `gsrc` to LDS `lds_dst`.  D# built per CDNA5 ISA ch.8 (group0/group1).
// This toolchain exposes the 6-arg builtin:
//   (uint32x4 g0, int32x8 g1, int32x4 g2, int32x4 g3, int32x8 extra, i32 cpol)
static __device__ __forceinline__ void tdm_load_edge_tile(const float* gsrc,
                                                          void* lds_dst) {
  const unsigned long long ga = (unsigned long long)(uintptr_t)gsrc;
  const unsigned int lds = (unsigned int)(uintptr_t)lds_dst;  // low32 = LDS off
  v4u_ g0;
  g0[0] = 1u;                                         // count=1 (user D#)
  g0[1] = lds;                                        // lds_addr
  g0[2] = (unsigned int)(ga & 0xFFFFFFFFu);           // global_addr[31:0]
  g0[3] = (unsigned int)((ga >> 32) & 0x1FFFFFFu)     // global_addr[56:32]
          | (2u << 30);                               // type = 2 ("image")
  v8i_ g1;
  g1[0] = (int)(2u << 16);                            // data_size = 4 bytes
  g1[1] = 0;                                          // tensor_dim0 = 65536 (lo16=0)
  g1[2] = (int)((2048u << 16) | 1u);                  // dim0 hi16=1, tensor_dim1=2048
  g1[3] = (int)(1024u << 16);                         // tile_dim0 = 1024
  g1[4] = 16;                                         // tile_dim1 = 16
  g1[5] = 65536;                                      // tensor_dim0_stride lo32
  g1[6] = 0;                                          // stride hi / dim1_stride
  g1[7] = 0;
  const v4i_ z4 = {0, 0, 0, 0};                       // groups 2/3: unused (2-D)
  const v8i_ z8 = {0, 0, 0, 0, 0, 0, 0, 0};
  __builtin_amdgcn_tensor_load_to_lds(g0, g1, z4, z4, z8, 0);
}
#endif

// ---------------------------------------------------------------------------
// k0: weight prep. wqkvT[c][k] = bf16(w_qkv[k][c]) (1536x128);
//     woutT[c][k] = bf16(w_out[k][c])  (128x512).
// ---------------------------------------------------------------------------
__global__ void NodeAttn_k0_prep(const float* __restrict__ w_qkv,
                                 const float* __restrict__ w_out,
                                 __bf16* __restrict__ wqkvT,
                                 __bf16* __restrict__ woutT) {
  const int t = blockIdx.x * blockDim.x + threadIdx.x;
  const int stride = gridDim.x * blockDim.x;
  for (int idx = t; idx < QKV3 * DIMC; idx += stride) {
    int c = idx >> 7, k = idx & 127;
    wqkvT[idx] = (__bf16)w_qkv[(size_t)k * QKV3 + c];
  }
  for (int idx = t; idx < DIMC * INNERC; idx += stride) {
    int c = idx >> 9, k = idx & 511;
    woutT[idx] = (__bf16)w_out[(size_t)k * DIMC + c];
  }
}

// ---------------------------------------------------------------------------
// k1: node LayerNorm + QKV GEMM (WMMA bf16, f32 accum). SCALE folded into q.
//   qbuf[(h*N + i)*64 + d], kbuf[(h*N + j)*64 + d], vT[(h*64 + d)*N + j]
// Store branch is wave-uniform (on n0), no exec divergence.
// ---------------------------------------------------------------------------
__global__ __launch_bounds__(256)
void NodeAttn_k1_ln_qkv(const float* __restrict__ nodes,
                        const float* __restrict__ node_g,
                        const float* __restrict__ node_b,
                        const __bf16* __restrict__ wqkvT,
                        __bf16* __restrict__ qbuf,
                        __bf16* __restrict__ kbuf,
                        __bf16* __restrict__ vT) {
  __shared__ __align__(16) float  xs[16][DIMC];
  __shared__ __align__(16) __bf16 xb[16][DIMC];
  __shared__ float mu_s[16], rs_s[16];

  const int tid = threadIdx.x;
  const int i0 = blockIdx.x * 16;

  {
    const float4* src = (const float4*)(nodes + (size_t)i0 * DIMC);
    float4* dst = (float4*)&xs[0][0];
    for (int t = tid; t < 16 * DIMC / 4; t += 256) dst[t] = src[t];
  }
  __syncthreads();
  if (tid < 16) {
    float s = 0.f, s2 = 0.f;
    for (int c = 0; c < DIMC; ++c) { float v = xs[tid][c]; s += v; s2 += v * v; }
    float mu = s * (1.f / DIMC);
    float var = s2 * (1.f / DIMC) - mu * mu;
    mu_s[tid] = mu; rs_s[tid] = rsqrtf(var + EPS_F);
  }
  __syncthreads();
  for (int t = tid; t < 16 * DIMC; t += 256) {
    int r = t >> 7, c = t & 127;
    xb[r][c] = (__bf16)((xs[r][c] - mu_s[r]) * rs_s[r] * node_g[c] + node_b[c]);
  }
  __syncthreads();

  const int wave = tid >> 5, lane = tid & 31;
  const int le = lane & 15, hi = lane >> 4;

  for (int t = wave; t < 96; t += 8) {
    const int n0 = t * 16;
    v8f acc = {0.f, 0.f, 0.f, 0.f, 0.f, 0.f, 0.f, 0.f};
#pragma unroll
    for (int kc = 0; kc < 4; ++kc) {            // K = 128 in 4 WMMA steps
      const int kb = kc * 32 + hi * 8;
      v16bf a = mk16(ld8(&xb[le][kb]), ld8(&xb[le][kb + 16]));
      const __bf16* bp = wqkvT + (size_t)(n0 + le) * DIMC + kb;
      v16bf b = mk16(ld8(bp), ld8(bp + 16));
      acc = wmma_bf16(a, b, acc);
    }
    // wave-uniform region select: tiles never cross q/k/v or head boundaries
    const int reg = n0 >> 9;            // 0=q, 1=k, 2=v
    const int hh  = (n0 >> 6) & 7;      // head
    const int d0  = (n0 & 63) + le;     // d within head
    if (reg == 0) {
#pragma unroll
      for (int r = 0; r < 8; ++r) {
        const int i = i0 + r + hi * 8;
        qbuf[((size_t)(hh * N_TOK + i)) * DHC + d0] = (__bf16)(acc[r] * SCALE_F);
      }
    } else if (reg == 1) {
#pragma unroll
      for (int r = 0; r < 8; ++r) {
        const int i = i0 + r + hi * 8;
        kbuf[((size_t)(hh * N_TOK + i)) * DHC + d0] = (__bf16)acc[r];
      }
    } else {
#pragma unroll
      for (int r = 0; r < 8; ++r) {
        const int i = i0 + r + hi * 8;
        vT[((size_t)(hh * DHC + d0)) * N_TOK + i] = (__bf16)acc[r];
      }
    }
  }
}

// ---------------------------------------------------------------------------
// k2: fused edge-LN + bias + masked flash attention.
// 256 thr = 8 waves (one head each) per 16-row i-block.  Edge tiles
// (16 x 32j x 32e f32 = 64KB) stream through double-buffered dynamic LDS via
// the Tensor Data Mover (wave 0 issues tensor_load_to_lds, retires with
// s_wait_tensorcnt); bias math in VALU co-executes with QK^T / PV WMMAs.
// ---------------------------------------------------------------------------
extern __shared__ __align__(16) char na_dyn_smem[];

__global__ __launch_bounds__(256)
void NodeAttn_k2_attn(const float* __restrict__ edges,
                      const unsigned char* __restrict__ adj,   // bool mask
                      const float* __restrict__ w_bias,        // (32,8)
                      const float* __restrict__ edge_g,
                      const float* __restrict__ edge_b,
                      const __bf16* __restrict__ qbuf,
                      const __bf16* __restrict__ kbuf,
                      const __bf16* __restrict__ vT,
                      __bf16* __restrict__ aout) {
  __shared__ __align__(16) float  bias_s[NH][16][32];   // 16 KB
  __shared__ __align__(16) __bf16 p_s[NH][16][32];      // 8 KB
  __shared__ __align__(16) float  wb_s[EDIMC][NH];      // 1 KB
  __shared__ unsigned char m_s[16][32];                 // 0.5 KB
  __shared__ float eg_s[EDIMC], eb_s[EDIMC];

  float* ebuf0 = (float*)na_dyn_smem;                   // 2 x 64 KB tiles
  float* ebuf1 = ebuf0 + 16 * 1024;

  const int tid = threadIdx.x;
  const int i0 = blockIdx.x * 16;
  const int wave = tid >> 5, lane = tid & 31;
  const int le = lane & 15, hi = lane >> 4;
  const int h = wave;
  const int NB = N_TOK / 32;

  for (int t = tid; t < EDIMC * NH; t += 256) wb_s[t >> 3][t & 7] = w_bias[t];
  if (tid < EDIMC) { eg_s[tid] = edge_g[tid]; eb_s[tid] = edge_b[tid]; }

  // Preload q fragments: A = 16x64 bf16, K split 2x32 (SCALE already folded).
  v16bf qa0, qa1;
  {
    const __bf16* qp = qbuf + ((size_t)(h * N_TOK + i0 + le)) * DHC;
    const int kb = hi * 8;
    qa0 = mk16(ld8(qp + kb),      ld8(qp + kb + 16));
    qa1 = mk16(ld8(qp + 32 + kb), ld8(qp + 32 + kb + 16));
  }

  v8f acc0 = {0,0,0,0,0,0,0,0}, acc1 = acc0, acc2 = acc0, acc3 = acc0;
  float mrow[8], lrow[8];
#pragma unroll
  for (int r = 0; r < 8; ++r) { mrow[r] = -1e38f; lrow[r] = 0.f; }

#if NA_USE_TDM
  if (wave == 0)   // prologue: DMA tile 0 into ebuf0
    tdm_load_edge_tile(edges + (size_t)i0 * (N_TOK * EDIMC), ebuf0);
#endif
  __syncthreads();  // wb_s/eg_s visible

  for (int jb = 0; jb < NB; ++jb) {
    const int j0 = jb * 32;
    float* cur = (jb & 1) ? ebuf1 : ebuf0;
#if NA_USE_TDM
    float* nxt = (jb & 1) ? ebuf0 : ebuf1;
    if (wave == 0) {
      if (jb + 1 < NB) {
        tdm_load_edge_tile(
            edges + ((size_t)i0 * N_TOK + (size_t)(jb + 1) * 32) * EDIMC, nxt);
        __builtin_amdgcn_s_wait_tensorcnt((short)1);  // tile jb landed
      } else {
        __builtin_amdgcn_s_wait_tensorcnt((short)0);  // last tile landed
      }
    }
#else
    // fallback: cooperative global->LDS stage of the 64KB tile
    for (int t = tid; t < 4096; t += 256) {
      const int row = t >> 8, c4 = t & 255;
      ((float4*)cur)[t] = ((const float4*)(
          edges + ((size_t)(i0 + row) * N_TOK + j0) * EDIMC))[c4];
    }
#endif
    __syncthreads();  // B0: edge tile `cur` visible to all waves

    // --- (1) bias + mask: 2 (i,j) pairs per thread, edge rows from LDS ----
    for (int p = tid; p < 512; p += 256) {
      const int ii = p >> 5, jj = p & 31;
      const float* ep = cur + (ii << 10) + (jj << 5);
      float ev[EDIMC];
#pragma unroll
      for (int q4 = 0; q4 < 8; ++q4)
        ((float4*)ev)[q4] = ((const float4*)ep)[q4];
      float s = 0.f, s2 = 0.f;
#pragma unroll
      for (int e = 0; e < EDIMC; ++e) { s += ev[e]; s2 += ev[e] * ev[e]; }
      const float mu = s * (1.f / EDIMC);
      const float rstd = rsqrtf(s2 * (1.f / EDIMC) - mu * mu + EPS_F);
      float ah[NH];
#pragma unroll
      for (int k = 0; k < NH; ++k) ah[k] = 0.f;
#pragma unroll
      for (int e = 0; e < EDIMC; ++e) {
        const float tv = (ev[e] - mu) * rstd * eg_s[e] + eb_s[e];
        const float4 w0 = ((const float4*)&wb_s[e][0])[0];
        const float4 w1 = ((const float4*)&wb_s[e][0])[1];
        ah[0] += tv * w0.x; ah[1] += tv * w0.y; ah[2] += tv * w0.z; ah[3] += tv * w0.w;
        ah[4] += tv * w1.x; ah[5] += tv * w1.y; ah[6] += tv * w1.z; ah[7] += tv * w1.w;
      }
#pragma unroll
      for (int k = 0; k < NH; ++k) bias_s[k][ii][jj] = ah[k];
      m_s[ii][jj] = adj[(size_t)(i0 + ii) * N_TOK + j0 + jj];
    }
    __syncthreads();  // B1: bias/mask visible

    // --- (2) QK^T: two 16x16 sim tiles, bias(+mask) as WMMA C -------------
    v8f sim[2];
#pragma unroll
    for (int jt = 0; jt < 2; ++jt) {
      v8f c;
#pragma unroll
      for (int r = 0; r < 8; ++r) {
        const int row = r + hi * 8;
        const int col = jt * 16 + le;
        c[r] = m_s[row][col] ? bias_s[h][row][col] : -1e30f;
      }
      const __bf16* kp = kbuf + ((size_t)(h * N_TOK + j0 + jt * 16 + le)) * DHC;
      const int kb = hi * 8;
      v16bf b0 = mk16(ld8(kp + kb),      ld8(kp + kb + 16));
      v16bf b1 = mk16(ld8(kp + 32 + kb), ld8(kp + 32 + kb + 16));
      c = wmma_bf16(qa0, b0, c);
      c = wmma_bf16(qa1, b1, c);
      sim[jt] = c;
    }

    // --- (3) online softmax (rows split across 16-lane halves) ------------
#pragma unroll
    for (int r = 0; r < 8; ++r) {
      float t = fmaxf(sim[0][r], sim[1][r]);
      t = fmaxf(t, __shfl_xor(t, 1, 32));
      t = fmaxf(t, __shfl_xor(t, 2, 32));
      t = fmaxf(t, __shfl_xor(t, 4, 32));
      t = fmaxf(t, __shfl_xor(t, 8, 32));
      const float mn = fmaxf(mrow[r], t);
      const float alpha = __expf(mrow[r] - mn);
      mrow[r] = mn;
      const float p0 = __expf(sim[0][r] - mn);
      const float p1 = __expf(sim[1][r] - mn);
      float rs = p0 + p1;
      rs += __shfl_xor(rs, 1, 32);
      rs += __shfl_xor(rs, 2, 32);
      rs += __shfl_xor(rs, 4, 32);
      rs += __shfl_xor(rs, 8, 32);
      lrow[r] = lrow[r] * alpha + rs;
      acc0[r] *= alpha; acc1[r] *= alpha; acc2[r] *= alpha; acc3[r] *= alpha;
      const int row = r + hi * 8;
      p_s[h][row][le]      = (__bf16)p0;   // transpose C-layout -> A-layout
      p_s[h][row][16 + le] = (__bf16)p1;
    }
    __syncthreads();  // B2: P visible

    // --- (4) PV: A = P (16x32), B = V tiles (32 x 16-d chunks) ------------
    {
      const int kb = hi * 8;
      const __bf16* pr = &p_s[h][le][0];
      v16bf a = mk16(ld8(pr + kb), ld8(pr + kb + 16));
#pragma unroll
      for (int nt = 0; nt < 4; ++nt) {
        const __bf16* vp = vT + ((size_t)(h * DHC + nt * 16 + le)) * N_TOK
                              + j0 + hi * 8;
        v16bf b = mk16(ld8(vp), ld8(vp + 16));
        v8f& ac = (nt == 0) ? acc0 : (nt == 1) ? acc1 : (nt == 2) ? acc2 : acc3;
        ac = wmma_bf16(a, b, ac);
      }
    }
  }

  // --- epilogue: normalize and emit bf16 (i, h*64+d) for final GEMM -------
#pragma unroll
  for (int r = 0; r < 8; ++r) {
    const int i = i0 + r + hi * 8;
    const float inv = 1.0f / lrow[r];
    __bf16* op = aout + (size_t)i * INNERC + h * DHC + le;
    op[0]  = (__bf16)(acc0[r] * inv);
    op[16] = (__bf16)(acc1[r] * inv);
    op[32] = (__bf16)(acc2[r] * inv);
    op[48] = (__bf16)(acc3[r] * inv);
  }
}

// ---------------------------------------------------------------------------
// k3: out = attn_out(2048x512) @ w_out(512x128) + b_out, f32 result.
// ---------------------------------------------------------------------------
__global__ __launch_bounds__(32)
void NodeAttn_k3_proj(const __bf16* __restrict__ aout,
                      const __bf16* __restrict__ woutT,
                      const float* __restrict__ b_out,
                      float* __restrict__ out) {
  const int bx = blockIdx.x;
  const int ib = bx >> 3, nt = bx & 7;
  const int i0 = ib * 16, n0 = nt * 16;
  const int lane = threadIdx.x & 31, le = lane & 15, hi = lane >> 4;

  v8f c;
  const float bb = b_out[n0 + le];
#pragma unroll
  for (int r = 0; r < 8; ++r) c[r] = bb;

#pragma unroll 4
  for (int kc = 0; kc < 16; ++kc) {           // K = 512
    const int kb = kc * 32 + hi * 8;
    const __bf16* ap = aout + (size_t)(i0 + le) * INNERC + kb;
    v16bf a = mk16(ld8(ap), ld8(ap + 16));
    const __bf16* bp = woutT + (size_t)(n0 + le) * INNERC + kb;
    v16bf b = mk16(ld8(bp), ld8(bp + 16));
    c = wmma_bf16(a, b, c);
  }
#pragma unroll
  for (int r = 0; r < 8; ++r)
    out[(size_t)(i0 + r + hi * 8) * DIMC + n0 + le] = c[r];
}

// ---------------------------------------------------------------------------
// Host launcher. d_in: nodes, edges, adj_mask, w_qkv, w_bias, w_out, b_out,
// node_g, node_b, edge_g, edge_b.  Workspace ~8.5 MB, all bf16.
// ---------------------------------------------------------------------------
extern "C" void kernel_launch(void* const* d_in, const int* in_sizes, int n_in,
                              void* d_out, int out_size, void* d_ws, size_t ws_size,
                              hipStream_t stream) {
  (void)in_sizes; (void)n_in; (void)out_size; (void)ws_size;
  const float* nodes  = (const float*)d_in[0];
  const float* edges  = (const float*)d_in[1];
  const unsigned char* adj = (const unsigned char*)d_in[2];  // bool mask
  const float* w_qkv  = (const float*)d_in[3];
  const float* w_bias = (const float*)d_in[4];
  const float* w_out  = (const float*)d_in[5];
  const float* b_out  = (const float*)d_in[6];
  const float* node_g = (const float*)d_in[7];
  const float* node_b = (const float*)d_in[8];
  const float* edge_g = (const float*)d_in[9];
  const float* edge_b = (const float*)d_in[10];
  float* out = (float*)d_out;

  char* ws = (char*)d_ws;
  __bf16* wqkvT = (__bf16*)(ws);                              // 1536*128*2
  __bf16* woutT = (__bf16*)(ws + 393216);                     // 128*512*2
  __bf16* qbuf  = (__bf16*)(ws + 524288);                     // 8*2048*64*2
  __bf16* kbuf  = (__bf16*)(ws + 524288 + 2097152);
  __bf16* vT    = (__bf16*)(ws + 524288 + 2u * 2097152);
  __bf16* aout  = (__bf16*)(ws + 524288 + 3u * 2097152);

  NodeAttn_k0_prep<<<dim3(256), dim3(256), 0, stream>>>(w_qkv, w_out, wqkvT, woutT);
  NodeAttn_k1_ln_qkv<<<dim3(N_TOK / 16), dim3(256), 0, stream>>>(
      nodes, node_g, node_b, wqkvT, qbuf, kbuf, vT);
  NodeAttn_k2_attn<<<dim3(N_TOK / 16), dim3(256), 131072, stream>>>(
      edges, adj, w_bias, edge_g, edge_b, qbuf, kbuf, vT, aout);
  NodeAttn_k3_proj<<<dim3((N_TOK / 16) * (DIMC / 16)), dim3(32), 0, stream>>>(
      aout, woutT, b_out, out);
}
